// UTSPGNNModel_64175401337461
// MI455X (gfx1250) — matrix operand order repaired and checked by reading, not compiled, not run on hardware
//
#include <hip/hip_runtime.h>
#include <hip/hip_bf16.h>
#include <stdint.h>

// GAT forward for B=16, N=2000, D=128, O=64, L=3 on gfx1250 (wave32, WMMA).
// Dominant op att@Wh done flash-style with v_wmma_f32_16x16x32_f16.

typedef __attribute__((ext_vector_type(16))) _Float16 v16h;
typedef __attribute__((ext_vector_type(8)))  float    v8f;

static constexpr int BB   = 16;
static constexpr int NN   = 2000;
static constexpr int DD   = 128;
static constexpr int OO   = 64;
static constexpr int NPAD = 2016;          // 63 * 32, zero-padded K dim

__device__ __forceinline__ float leaky02(float x) { return x >= 0.f ? x : 0.2f * x; }

// ---------------- weight convert + transpose: W[din,dout] f32 -> WT[dout,din] f16
__global__ void k_wt16(const float* __restrict__ src, _Float16* __restrict__ dst,
                       int din, int dout) {
    int i = blockIdx.x * blockDim.x + threadIdx.x;
    if (i < din * dout) {
        int o = i % dout, k = i / dout;
        dst[o * din + k] = (_Float16)src[i];
    }
}

// ---------------- encoder layer 1: hid = relu(coords @ w0 + b0), K=2 (memory bound)
__global__ void k_enc_hid(const float* __restrict__ coords, const float* __restrict__ w0,
                          const float* __restrict__ b0, float* __restrict__ hid) {
    int i = blockIdx.x * blockDim.x + threadIdx.x;      // over B*N*D
    if (i < BB * NN * DD) {
        int d = i & (DD - 1);
        int node = i >> 7;
        float c0 = coords[node * 2 + 0], c1 = coords[node * 2 + 1];
        float v = fmaf(c0, w0[d], fmaf(c1, w0[DD + d], b0[d]));
        hid[i] = v > 0.f ? v : 0.f;
    }
}

// ---------------- generic WMMA linear: Y[M,dout] = X[M,128] @ WT^T (+bias)
// block = 16 rows x dout cols; blockDim = 32 * (dout/16); WT is [dout][128] f16.
__global__ void k_linear(const float* __restrict__ X, const _Float16* __restrict__ WT,
                         const float* __restrict__ bias, float* __restrict__ Y, int dout) {
    __shared__ __align__(16) uint32_t As[4 * 320];      // 4 K-chunks x 16 rows x 20 dwords
    const int tid = threadIdx.x;
    const int row0 = blockIdx.x * 16;
    // stage A tile (16 x 128) as f16 pairs in WMMA A-operand order
    for (int p = tid; p < 1024; p += blockDim.x) {
        int c = p >> 8, rem = p & 255;
        int row = rem >> 4, pp = rem & 15;
        int k = c * 32 + pp * 2;
        const float* xr = X + (size_t)(row0 + row) * DD + k;
        union { _Float16 h[2]; uint32_t u; } pk;
        pk.h[0] = (_Float16)xr[0];
        pk.h[1] = (_Float16)xr[1];
        As[c * 320 + row * 20 + pp] = pk.u;
    }
    __syncthreads();

    const int wave = tid >> 5, lane = tid & 31;
    const int rowA = lane & 15, hs = lane >> 4;
    const int colg = wave * 16 + (lane & 15);
    v8f acc = {};
#pragma unroll
    for (int cc = 0; cc < 4; ++cc) {
        union { uint4 q[2]; v16h v; } av, bv;
        const uint32_t* ap = &As[cc * 320 + rowA * 20 + hs * 4];
        av.q[0] = *(const uint4*)ap;
        av.q[1] = *(const uint4*)(ap + 8);
        const _Float16* bp = WT + (size_t)colg * DD + cc * 32 + hs * 8;
        bv.q[0] = *(const uint4*)bp;
        bv.q[1] = *(const uint4*)(bp + 16);
        acc = __builtin_amdgcn_wmma_f32_16x16x32_f16(false, av.v, false, bv.v,
                                                     (short)0, acc, false, false);
    }
    float bcol = bias ? bias[colg] : 0.f;
    const int mbase = hs * 8;
#pragma unroll
    for (int r = 0; r < 8; ++r) {
        int m = mbase + r;
        Y[(size_t)(row0 + m) * dout + colg] = acc[r] + bcol;
    }
}

// ---------------- per-layer prep: WhT[b][d][j] f16 (zero-padded), e1/e2 reductions
__global__ void k_prep(const float* __restrict__ Wh, const float* __restrict__ a1,
                       const float* __restrict__ a2, _Float16* __restrict__ WhT,
                       float* __restrict__ e1, float* __restrict__ e2) {
    __shared__ float red[DD];
    const int node = blockIdx.x, b = blockIdx.y, d = threadIdx.x;
    _Float16* wt = WhT + ((size_t)b * DD + d) * NPAD + node;
    if (node >= NN) { *wt = (_Float16)0.f; return; }     // uniform per block
    float w = Wh[((size_t)b * NN + node) * DD + d];
    *wt = (_Float16)w;
    red[d] = w * a1[d];
    __syncthreads();
    for (int s = 64; s > 0; s >>= 1) { if (d < s) red[d] += red[d + s]; __syncthreads(); }
    if (d == 0) e1[b * NN + node] = red[0];
    __syncthreads();
    red[d] = w * a2[d];
    __syncthreads();
    for (int s = 64; s > 0; s >>= 1) { if (d < s) red[d] += red[d + s]; __syncthreads(); }
    if (d == 0) e2[b * NN + node] = red[0];
}

// ---------------- per-batch max_j e2 (for the softmax row-max identity)
__global__ void k_maxe2(const float* __restrict__ e2, float* __restrict__ Mb) {
    __shared__ float red[256];
    const int b = blockIdx.x, t = threadIdx.x;
    float m = -3.4e38f;
    for (int j = t; j < NN; j += 256) m = fmaxf(m, e2[b * NN + j]);
    red[t] = m; __syncthreads();
    for (int s = 128; s > 0; s >>= 1) { if (t < s) red[t] = fmaxf(red[t], red[t + s]); __syncthreads(); }
    if (t == 0) Mb[b] = red[0];
}

// ---------------- fused softmax-attention GEMM: H = relu(softmax(e) @ Wh)
// grid (125, 16); block 256 = 8 waves; wave w owns output cols [16w,16w+16).
__global__ void k_attn(const _Float16* __restrict__ WhT, const float* __restrict__ e1,
                       const float* __restrict__ e2, const float* __restrict__ Mb,
                       float* __restrict__ Hout) {
    constexpr int SUB = 3;                                // 96 j per stage; 2016/96 = 21
    __shared__ __align__(16) uint32_t As[SUB * 320];
    __shared__ float e1s[16], ms[16];
    __shared__ float Zp[256];
    __shared__ float Zrow[16];
    const int tid = threadIdx.x;
    const int row0 = blockIdx.x * 16;
    const int b = blockIdx.y;
    if (tid < 16) {
        float ev = e1[b * NN + row0 + tid];
        e1s[tid] = ev;
        ms[tid] = leaky02(ev + Mb[b]);                    // leaky monotone => exact row max
    }
    __syncthreads();

    const int wave = tid >> 5, lane = tid & 31;
    const int rowA = lane & 15, hs = lane >> 4;
    const int colg = wave * 16 + (lane & 15);
    const _Float16* Wbase = WhT + ((size_t)b * DD + colg) * NPAD;
    const int wrow = tid >> 4, wp = tid & 15;
    const float e1r = e1s[wrow], mr = ms[wrow];
    const float* e2b = e2 + b * NN;
    float zsum = 0.f;
    v8f acc = {};

    for (int jb = 0; jb < NPAD; jb += 32 * SUB) {
#pragma unroll
        for (int s = 0; s < SUB; ++s) {                   // 256 thr compute 16x96 weights once
            int j = jb + s * 32 + wp * 2;
            float w0 = 0.f, w1 = 0.f;
            if (j < NN)     w0 = __expf(leaky02(e1r + e2b[j])     - mr);
            if (j + 1 < NN) w1 = __expf(leaky02(e1r + e2b[j + 1]) - mr);
            union { _Float16 h[2]; uint32_t u; } pk;
            pk.h[0] = (_Float16)w0; pk.h[1] = (_Float16)w1;
            zsum += (float)pk.h[0] + (float)pk.h[1];      // Z from the f16-rounded weights
            As[s * 320 + wrow * 20 + wp] = pk.u;
        }
        __syncthreads();
#pragma unroll
        for (int s = 0; s < SUB; ++s) {
            union { uint4 q[2]; v16h v; } av, bv;
            const uint32_t* ap = &As[s * 320 + rowA * 20 + hs * 4];
            av.q[0] = *(const uint4*)ap;
            av.q[1] = *(const uint4*)(ap + 8);
            const _Float16* bp = Wbase + jb + s * 32 + hs * 8;  // L2-resident
            bv.q[0] = *(const uint4*)bp;
            bv.q[1] = *(const uint4*)(bp + 16);
            acc = __builtin_amdgcn_wmma_f32_16x16x32_f16(false, av.v, false, bv.v,
                                                         (short)0, acc, false, false);
        }
        __syncthreads();
    }

    Zp[tid] = zsum; __syncthreads();
    if (tid < 16) {
        float s = 0.f;
        for (int p = 0; p < 16; ++p) s += Zp[tid * 16 + p];
        Zrow[tid] = s;
    }
    __syncthreads();
    const int mbase = hs * 8;
#pragma unroll
    for (int r = 0; r < 8; ++r) {
        int m = mbase + r;
        float v = acc[r];
        v = v > 0.f ? v : 0.f;
        Hout[((size_t)b * NN + row0 + m) * DD + colg] = v / Zrow[m];
    }
}

extern "C" void kernel_launch(void* const* d_in, const int* in_sizes, int n_in,
                              void* d_out, int out_size, void* d_ws, size_t ws_size,
                              hipStream_t stream) {
    const float* coords = (const float*)d_in[0];
    const float* enc_w0 = (const float*)d_in[1];
    const float* enc_b0 = (const float*)d_in[2];
    const float* enc_w1 = (const float*)d_in[3];
    const float* enc_b1 = (const float*)d_in[4];
    const float* gat_W  = (const float*)d_in[5];
    const float* gat_a1 = (const float*)d_in[6];
    const float* gat_a2 = (const float*)d_in[7];
    const float* proj_w = (const float*)d_in[8];
    const float* proj_b = (const float*)d_in[9];
    float* out = (float*)d_out;

    char* ws = (char*)d_ws;
    auto carve = [&](size_t bytes) -> char* {
        char* p = ws; ws += (bytes + 255) & ~(size_t)255; return p;
    };
    float*     bufA = (float*)    carve(sizeof(float) * BB * NN * DD);   // hid / Wh
    float*     bufH = (float*)    carve(sizeof(float) * BB * NN * DD);   // h
    _Float16*  WhT  = (_Float16*) carve(sizeof(_Float16) * BB * DD * NPAD);
    _Float16*  w1T  = (_Float16*) carve(sizeof(_Float16) * DD * DD);
    _Float16*  gWT  = (_Float16*) carve(sizeof(_Float16) * 3 * DD * DD);
    _Float16*  pWT  = (_Float16*) carve(sizeof(_Float16) * OO * DD);
    float*     e1   = (float*)    carve(sizeof(float) * BB * NN);
    float*     e2   = (float*)    carve(sizeof(float) * BB * NN);
    float*     Mb   = (float*)    carve(sizeof(float) * BB);

    // weight conversions (f32 -> transposed f16 B-operand layout)
    k_wt16<<<(DD * DD + 255) / 256, 256, 0, stream>>>(enc_w1, w1T, DD, DD);
    for (int l = 0; l < 3; ++l)
        k_wt16<<<(DD * DD + 255) / 256, 256, 0, stream>>>(gat_W + l * DD * DD,
                                                          gWT + l * DD * DD, DD, DD);
    k_wt16<<<(DD * OO + 255) / 256, 256, 0, stream>>>(proj_w, pWT, DD, OO);

    // encoder
    k_enc_hid<<<(BB * NN * DD) / 256, 256, 0, stream>>>(coords, enc_w0, enc_b0, bufA);
    k_linear<<<(BB * NN) / 16, 256, 0, stream>>>(bufA, w1T, enc_b1, bufH, DD);

    // GAT layers
    for (int l = 0; l < 3; ++l) {
        k_linear<<<(BB * NN) / 16, 256, 0, stream>>>(bufH, gWT + l * DD * DD,
                                                     nullptr, bufA, DD);       // Wh
        k_prep<<<dim3(NPAD, BB), DD, 0, stream>>>(bufA, gat_a1 + l * DD,
                                                  gat_a2 + l * DD, WhT, e1, e2);
        k_maxe2<<<BB, 256, 0, stream>>>(e2, Mb);
        k_attn<<<dim3(NN / 16, BB), 256, 0, stream>>>(WhT, e1, e2, Mb, bufH);
    }

    // projection head (dout = 64 -> 4 waves per block)
    k_linear<<<(BB * NN) / 16, 128, 0, stream>>>(bufH, pWT, proj_b, out, OO);
}
